// Decoder_17549236371679
// MI455X (gfx1250) — matrix operand order repaired and checked by reading, not compiled, not run on hardware
//
#include <hip/hip_runtime.h>
#include <hip/hip_bf16.h>
#include <math.h>

// ---- problem dims (fixed by the reference) ----
#define Vv 32000
#define Ee 512
#define Hh 1024
#define Ss 256
#define Bb 128

typedef __attribute__((ext_vector_type(16))) __bf16 v16bf;
typedef __attribute__((ext_vector_type(8)))  float  v8f;
typedef __attribute__((ext_vector_type(4)))  int    v4i;

#define ASG __attribute__((address_space(1)))
#define ASL __attribute__((address_space(3)))
typedef ASG v4i* gv4i_p;   // global 128-bit chunk pointer
typedef ASL v4i* lv4i_p;   // LDS    128-bit chunk pointer

__device__ __forceinline__ v16bf pack16(float4 x0, float4 x1, float4 x2, float4 x3)
{
    v16bf r;
    r[0]=(__bf16)x0.x;  r[1]=(__bf16)x0.y;  r[2]=(__bf16)x0.z;  r[3]=(__bf16)x0.w;
    r[4]=(__bf16)x1.x;  r[5]=(__bf16)x1.y;  r[6]=(__bf16)x1.z;  r[7]=(__bf16)x1.w;
    r[8]=(__bf16)x2.x;  r[9]=(__bf16)x2.y;  r[10]=(__bf16)x2.z; r[11]=(__bf16)x2.w;
    r[12]=(__bf16)x3.x; r[13]=(__bf16)x3.y; r[14]=(__bf16)x3.z; r[15]=(__bf16)x3.w;
    return r;
}

// ------------------------------------------------------------------
// out[M,N] = X[M,K] @ W[N,K]^T + bias[N]  via bf16 WMMA, f32 accumulate.
// blockDim = 128 (4 waves); each wave owns one 16x16 output tile.
// grid = (N/64, M/16). K must be a multiple of 64.
// A tile (16 x 64 f32 chunk) is staged block-wide into LDS with
// GLOBAL_LOAD_ASYNC_TO_LDS_B128, double-buffered (ASYNCcnt pipelining).
// W rows are streamed straight from global (read exactly once).
// ------------------------------------------------------------------
__global__ __launch_bounds__(128)
void gemm_bf16_wmma(const float* __restrict__ X, const float* __restrict__ W,
                    const float* __restrict__ bias, float* __restrict__ out,
                    int M, int N, int K)
{
    constexpr int APAD = 68;                 // 64 + 4 pad floats -> bank spread
    __shared__ float Atile[2][16 * APAD];

    const int tid  = threadIdx.x;
    const int lane = tid & 31;
    const int wave = tid >> 5;
    const int half = lane >> 4;              // 0: lanes 0-15, 1: lanes 16-31
    const int l16  = lane & 15;

    const int nbase = (blockIdx.x * 4 + wave) * 16;
    const int mbase = blockIdx.y * 16;

    const float* __restrict__ wrow = W + (size_t)(nbase + l16) * K;

    // Staging map: linear 16B-chunk id q in [0,256): row = q>>4, col4 = (q&15)*4.
    const int q0 = tid * 2;
    const int r0 = q0 >> 4,       c0 = (q0 & 15) * 4;
    const int r1 = (q0 + 1) >> 4, c1 = ((q0 + 1) & 15) * 4;
    const float* g0 = X + (size_t)(mbase + r0) * K + c0;
    const float* g1 = X + (size_t)(mbase + r1) * K + c1;

#define STAGE(buf_, kc_)                                                        \
    do {                                                                        \
        __builtin_amdgcn_global_load_async_to_lds_b128(                         \
            (gv4i_p)(g0 + (kc_)),                                               \
            (lv4i_p)&Atile[(buf_)][r0 * APAD + c0], 0, 0);                      \
        __builtin_amdgcn_global_load_async_to_lds_b128(                         \
            (gv4i_p)(g1 + (kc_)),                                               \
            (lv4i_p)&Atile[(buf_)][r1 * APAD + c1], 0, 0);                      \
    } while (0)

    v8f c = {};
    int buf = 0;
    STAGE(0, 0);

    for (int kc = 0; kc < K; kc += 64) {
        const bool more = (kc + 64) < K;
        if (more) STAGE(buf ^ 1, kc + 64);

        // ---- B fragments for both 32-wide K sub-chunks (direct global) ----
        __builtin_prefetch(wrow + kc + 256, 0, 1);   // global_prefetch_b8
        const float4* pb0 = reinterpret_cast<const float4*>(wrow + kc + half * 16);
        const float4* pb1 = reinterpret_cast<const float4*>(wrow + kc + 32 + half * 16);
        float4 b00 = pb0[0], b01 = pb0[1], b02 = pb0[2], b03 = pb0[3];
        float4 b10 = pb1[0], b11 = pb1[1], b12 = pb1[2], b13 = pb1[3];

        // previous stage (this buf) complete; next stage may stay in flight
        if (more) __builtin_amdgcn_s_wait_asynccnt(2);
        else      __builtin_amdgcn_s_wait_asynccnt(0);
        __syncthreads();

        // ---- A fragments from LDS (contiguous k = 16*half + s, same order as B) ----
        const float* abase = &Atile[buf][l16 * APAD];
        float4 a00 = *(const float4*)(abase + half * 16);
        float4 a01 = *(const float4*)(abase + half * 16 + 4);
        float4 a02 = *(const float4*)(abase + half * 16 + 8);
        float4 a03 = *(const float4*)(abase + half * 16 + 12);
        float4 a10 = *(const float4*)(abase + 32 + half * 16);
        float4 a11 = *(const float4*)(abase + 32 + half * 16 + 4);
        float4 a12 = *(const float4*)(abase + 32 + half * 16 + 8);
        float4 a13 = *(const float4*)(abase + 32 + half * 16 + 12);

        v16bf A0 = pack16(a00, a01, a02, a03);
        v16bf B0 = pack16(b00, b01, b02, b03);
        c = __builtin_amdgcn_wmma_f32_16x16x32_bf16(false, A0, false, B0,
                                                    (short)0, c, false, false);
        v16bf A1 = pack16(a10, a11, a12, a13);
        v16bf B1 = pack16(b10, b11, b12, b13);
        c = __builtin_amdgcn_wmma_f32_16x16x32_bf16(false, A1, false, B1,
                                                    (short)0, c, false, false);

        __syncthreads();   // protect buf before it is overwritten next iter
        buf ^= 1;
    }
#undef STAGE

    const int col = nbase + l16;
    const float bv = bias ? bias[col] : 0.0f;
    #pragma unroll
    for (int i = 0; i < 8; ++i) {
        const int row = mbase + i + 8 * half;
        out[(size_t)row * N + col] = c[i] + bv;
    }
}

// ------------------------------------------------------------------
// Embedding gather + per-batch encoder length (first s with mask==0, else S)
// ------------------------------------------------------------------
__global__ void k_prep(const int* __restrict__ x, const float* __restrict__ emb,
                       const int* __restrict__ mask,
                       float* __restrict__ xemb, float* __restrict__ lenf)
{
    const int b = blockIdx.x;
    const int row = x[b];
    for (int e = threadIdx.x; e < Ee; e += blockDim.x)
        xemb[(size_t)b * Ee + e] = emb[(size_t)row * Ee + e];
    if (threadIdx.x == 0) {
        int L = Ss;
        for (int s = 0; s < Ss; ++s) { if (mask[s * Bb + b] == 0) { L = s; break; } }
        lenf[b] = (float)L;
    }
}

// ------------------------------------------------------------------
// pt[b] = (len[b]-1) * sigmoid( sum_h tanh(tpos[b,h]) * wproj[h] )
// ------------------------------------------------------------------
__global__ void k_pt(const float* __restrict__ tpos, const float* __restrict__ wproj,
                     const float* __restrict__ lenf, float* __restrict__ pt)
{
    const int b = blockIdx.x, tid = threadIdx.x;
    __shared__ float red[256];
    float acc = 0.0f;
    for (int h = tid; h < Hh; h += 256)
        acc += tanhf(tpos[(size_t)b * Hh + h]) * wproj[h];
    red[tid] = acc; __syncthreads();
    for (int st = 128; st > 0; st >>= 1) {
        if (tid < st) red[tid] += red[tid + st];
        __syncthreads();
    }
    if (tid == 0) {
        const float f = 1.0f / (1.0f + expf(-red[0]));
        pt[b] = (lenf[b] - 1.0f) * f;
    }
}

// ------------------------------------------------------------------
// energies -> masked softmax over S -> Gaussian local window -> w(s,b)
// one block per batch element; blockDim = 256 (== S)
// ------------------------------------------------------------------
__global__ void k_attn(const float* __restrict__ dw, const float* __restrict__ enc,
                       const int* __restrict__ mask, const float* __restrict__ pt,
                       float* __restrict__ wout)
{
    const int b = blockIdx.x;
    const int tid = threadIdx.x, lane = tid & 31, wid = tid >> 5;
    __shared__ float sE[Ss];
    __shared__ float red[256];

    const float* __restrict__ drow = dw + (size_t)b * Hh;
    for (int s = wid; s < Ss; s += 8) {
        const float* __restrict__ erow = enc + ((size_t)s * Bb + b) * Hh;
        float acc = 0.0f;
        for (int h = lane; h < Hh; h += 32)
            acc += drow[h] * erow[h];
        for (int off = 16; off > 0; off >>= 1)
            acc += __shfl_down(acc, off, 32);
        if (lane == 0) sE[s] = tanhf(acc);
    }
    __syncthreads();

    const int m = mask[tid * Bb + b];
    const float e = sE[tid];
    red[tid] = m ? e : -INFINITY; __syncthreads();
    for (int st = 128; st > 0; st >>= 1) {
        if (tid < st) red[tid] = fmaxf(red[tid], red[tid + st]);
        __syncthreads();
    }
    const float mx = red[0]; __syncthreads();

    const float ex = m ? expf(e - mx) : 0.0f;
    red[tid] = ex; __syncthreads();
    for (int st = 128; st > 0; st >>= 1) {
        if (tid < st) red[tid] += red[tid + st];
        __syncthreads();
    }
    const float att = ex / red[0];

    const float ptb = pt[b];
    const float sf  = (float)tid;
    const float win = expf(-(ptb - sf) * (ptb - sf) * (1.0f / 12.5f)); // 2*(D/2)^2
    const float pmaxv = floorf(ptb + 5.0f);
    const float pminv = floorf(fmaxf(ptb - 5.0f, 0.0f));
    const float hard  = (sf < pmaxv && sf >= pminv) ? 1.0f : 0.0f;
    wout[tid * Bb + b] = att * win * hard;
}

// ------------------------------------------------------------------
// context[b,h] = sum_s w(s,b) * enc[s,b,h]   (second, L2-resident pass)
// ------------------------------------------------------------------
__global__ void k_context(const float* __restrict__ wgt, const float* __restrict__ enc,
                          float* __restrict__ ctx)
{
    const int b = blockIdx.y;
    const int h = blockIdx.x * blockDim.x + threadIdx.x;
    float acc = 0.0f;
    for (int s = 0; s < Ss; ++s)
        acc += wgt[s * Bb + b] * enc[((size_t)s * Bb + b) * Hh + h];
    ctx[(size_t)b * Hh + h] = acc;
}

// ------------------------------------------------------------------
__global__ void k_concat_gruin(const float* __restrict__ xemb, const float* __restrict__ ctx,
                               float* __restrict__ gin)
{
    const int idx = blockIdx.x * blockDim.x + threadIdx.x;
    const int K = Ee + Hh;
    const int b = idx / K, j = idx % K;
    gin[idx] = (j < Ee) ? xemb[(size_t)b * Ee + j] : ctx[(size_t)b * Hh + (j - Ee)];
}

// ------------------------------------------------------------------
__global__ void k_gru(const float* __restrict__ gi, const float* __restrict__ gh,
                      const float* __restrict__ phd, float* __restrict__ hnew,
                      float* __restrict__ out_tail)
{
    const int idx = blockIdx.x * blockDim.x + threadIdx.x;
    const int b = idx / Hh, h = idx % Hh;
    const size_t base = (size_t)b * 3 * Hh + h;
    const float r  = 1.0f / (1.0f + expf(-(gi[base] + gh[base])));
    const float zg = 1.0f / (1.0f + expf(-(gi[base + Hh] + gh[base + Hh])));
    const float n  = tanhf(gi[base + 2 * Hh] + r * gh[base + 2 * Hh]);
    const float hp = phd[(size_t)b * Hh + h];
    const float hn = (1.0f - zg) * n + zg * hp;
    hnew[(size_t)b * Hh + h] = hn;
    out_tail[(size_t)b * Hh + h] = hn;
}

// ------------------------------------------------------------------
__global__ void k_concat_fcin(const float* __restrict__ xemb, const float* __restrict__ hnew,
                              const float* __restrict__ ctx, float* __restrict__ fin)
{
    const int idx = blockIdx.x * blockDim.x + threadIdx.x;
    const int K = Ee + 2 * Hh;
    const int b = idx / K, j = idx % K;
    float v;
    if (j < Ee)            v = xemb[(size_t)b * Ee + j];
    else if (j < Ee + Hh)  v = hnew[(size_t)b * Hh + (j - Ee)];
    else                   v = ctx[(size_t)b * Hh + (j - Ee - Hh)];
    fin[idx] = v;
}

// ------------------------------------------------------------------
extern "C" void kernel_launch(void* const* d_in, const int* in_sizes, int n_in,
                              void* d_out, int out_size, void* d_ws, size_t ws_size,
                              hipStream_t stream)
{
    const int*   x    = (const int*)  d_in[0];
    const float* phd  = (const float*)d_in[1];   // (1,B,H) -> (B,H)
    const float* enc  = (const float*)d_in[2];   // (S,B,H)
    const int*   msk  = (const int*)  d_in[3];   // (S,B)
    const float* emb  = (const float*)d_in[4];   // (V,E)
    const float* Wdec = (const float*)d_in[5];
    const float* bdec = (const float*)d_in[6];
    const float* Wpos = (const float*)d_in[7];
    const float* bpos = (const float*)d_in[8];
    const float* wprj = (const float*)d_in[9];   // (1,H)
    const float* Wih  = (const float*)d_in[10];  // (3H, E+H)
    const float* bih  = (const float*)d_in[11];
    const float* Whh  = (const float*)d_in[12];  // (3H, H)
    const float* bhh  = (const float*)d_in[13];
    const float* Wfc  = (const float*)d_in[14];  // (V, E+2H)
    const float* bfc  = (const float*)d_in[15];

    float* out = (float*)d_out;                  // [B*V | B*H]
    float* ws  = (float*)d_ws;

    size_t o = 0;
    float* xemb = ws + o; o += (size_t)Bb * Ee;
    float* dw   = ws + o; o += (size_t)Bb * Hh;
    float* tpos = ws + o; o += (size_t)Bb * Hh;
    float* pt   = ws + o; o += Bb;
    float* lenf = ws + o; o += Bb;
    float* wgt  = ws + o; o += (size_t)Ss * Bb;
    float* ctx  = ws + o; o += (size_t)Bb * Hh;
    float* gin  = ws + o; o += (size_t)Bb * (Ee + Hh);
    float* gi   = ws + o; o += (size_t)Bb * 3 * Hh;
    float* gh   = ws + o; o += (size_t)Bb * 3 * Hh;
    float* hnew = ws + o; o += (size_t)Bb * Hh;
    float* fin  = ws + o; o += (size_t)Bb * (Ee + 2 * Hh);

    // 1) embedding gather + encoder lengths
    k_prep<<<Bb, 256, 0, stream>>>(x, emb, msk, xemb, lenf);

    // 2) dw = phd @ Wdec^T + bdec          (128 x 1024, K=1024)
    gemm_bf16_wmma<<<dim3(Hh / 64, Bb / 16), 128, 0, stream>>>(
        phd, Wdec, bdec, dw, Bb, Hh, Hh);

    // 3) tpos = phd @ Wpos^T + bpos
    gemm_bf16_wmma<<<dim3(Hh / 64, Bb / 16), 128, 0, stream>>>(
        phd, Wpos, bpos, tpos, Bb, Hh, Hh);

    // 4) pt
    k_pt<<<Bb, 256, 0, stream>>>(tpos, wprj, lenf, pt);

    // 5) attention weights w(s,b)
    k_attn<<<Bb, 256, 0, stream>>>(dw, enc, msk, pt, wgt);

    // 6) context
    k_context<<<dim3(Hh / 256, Bb), 256, 0, stream>>>(wgt, enc, ctx);

    // 7) gru_in
    k_concat_gruin<<<(Bb * (Ee + Hh)) / 256, 256, 0, stream>>>(xemb, ctx, gin);

    // 8) gi = gru_in @ Wih^T + bih         (128 x 3072, K=1536)
    gemm_bf16_wmma<<<dim3((3 * Hh) / 64, Bb / 16), 128, 0, stream>>>(
        gin, Wih, bih, gi, Bb, 3 * Hh, Ee + Hh);

    // 9) gh = phd @ Whh^T + bhh            (128 x 3072, K=1024)
    gemm_bf16_wmma<<<dim3((3 * Hh) / 64, Bb / 16), 128, 0, stream>>>(
        phd, Whh, bhh, gh, Bb, 3 * Hh, Hh);

    // 10) GRU gates + h_new (also output tail)
    k_gru<<<(Bb * Hh) / 256, 256, 0, stream>>>(gi, gh, phd, hnew, out + (size_t)Bb * Vv);

    // 11) fc_in
    k_concat_fcin<<<(Bb * (Ee + 2 * Hh)) / 256, 256, 0, stream>>>(xemb, hnew, ctx, fin);

    // 12) z = fc_in @ Wfc^T + bfc          (128 x 32000, K=2560) — dominant
    gemm_bf16_wmma<<<dim3(Vv / 64, Bb / 16), 128, 0, stream>>>(
        fin, Wfc, bfc, out, Bb, Vv, Ee + 2 * Hh);
}
